// MoeLayer_48000554500884
// MI455X (gfx1250) — matrix operand order repaired and checked by reading, not compiled, not run on hardware
//
#include <hip/hip_runtime.h>

// MoE (top-2 of 8 experts + shared expert), B*S=8192 tokens, D=1024, H=2048.
// Routed bf16-WMMA formulation: gate -> gather per expert -> per-expert
// fused SwiGLU GEMMs on v_wmma_f32_16x16x32_bf16, fp32 accumulation.
// Phase B stages its bf16 activation tiles with GLOBAL_LOAD_ASYNC_TO_LDS_B128
// (ASYNCcnt + s_wait_asynccnt), overlapping with fp32->bf16 weight conversion.
// Deterministic: no float atomics; expert contributions added in fixed
// launch order; within an expert each output element is touched once.

#define N_TOK   8192
#define D_IN    1024
#define H_DIM   2048
#define NEXP    8

#define LDT     40            // LDS row stride in bf16 elems (80B, 16B-aligned)

typedef __attribute__((ext_vector_type(16))) __bf16 v16bf;
typedef __attribute__((ext_vector_type(8)))  float  v8f;

// ---- fragment loads from LDS (wave32 layouts per CDNA5 ISA 7.12.2) ----
// A (16x32 bf16): lane L holds m=L&15; K pairs {kb..kb+7} and {kb+16..kb+23},
// kb = (L>=16)?8:0  -> two 16B chunks at +0 and +16 elements.
__device__ inline v16bf ld_frag_a(const __bf16* p) {
    union { v16bf v; float4 q[2]; } u;
    u.q[0] = *(const float4*)(p);
    u.q[1] = *(const float4*)(p + 16);
    return u.v;
}
// B (32x16 bf16): lane L holds n=L&15; K pairs {kb..kb+7} and {kb+16..kb+23},
// kb = (L>=16)?8:0  -> two 16B chunks at +0 and +16 elements.
__device__ inline v16bf ld_frag_b(const __bf16* p) {
    union { v16bf v; float4 q[2]; } u;
    u.q[0] = *(const float4*)(p);
    u.q[1] = *(const float4*)(p + 16);
    return u.v;
}

__device__ inline void store_arow(__bf16* d, float4 f0, float4 f1) {
    d[0]=(__bf16)f0.x; d[1]=(__bf16)f0.y; d[2]=(__bf16)f0.z; d[3]=(__bf16)f0.w;
    d[4]=(__bf16)f1.x; d[5]=(__bf16)f1.y; d[6]=(__bf16)f1.z; d[7]=(__bf16)f1.w;
}
__device__ inline void store_bcol(__bf16* d, float4 f0, float4 f1) {
    d[0*LDT]=(__bf16)f0.x; d[1*LDT]=(__bf16)f0.y;
    d[2*LDT]=(__bf16)f0.z; d[3*LDT]=(__bf16)f0.w;
    d[4*LDT]=(__bf16)f1.x; d[5*LDT]=(__bf16)f1.y;
    d[6*LDT]=(__bf16)f1.z; d[7*LDT]=(__bf16)f1.w;
}

// async 16B global->LDS copy (GLOBAL_LOAD_ASYNC_TO_LDS_B128, ASYNCcnt).
// LDS byte offset = low 32 bits of the generic pointer (ISA aperture rule).
__device__ inline void async_copy_b128(const void* gsrc, void* ldst) {
    const unsigned int  loff = (unsigned int)(unsigned long long)ldst;
    const unsigned long long ga = (unsigned long long)gsrc;
    asm volatile("global_load_async_to_lds_b128 %0, %1, off"
                 :: "v"(loff), "v"(ga) : "memory");
}
__device__ inline void wait_async0() {
    asm volatile("s_wait_asynccnt 0x0" ::: "memory");
}

// ---------------- gating: one wave per token ----------------
__global__ __launch_bounds__(256)
void moe_gate(const float* __restrict__ x, const float* __restrict__ gwm,
              int* __restrict__ tok_e, float* __restrict__ tok_w,
              int* __restrict__ counts)
{
    const int wave = threadIdx.x >> 5;
    const int lane = threadIdx.x & 31;
    const int t = blockIdx.x * 8 + wave;
    if (t >= N_TOK) return;

    float acc[NEXP];
#pragma unroll
    for (int e = 0; e < NEXP; ++e) acc[e] = 0.0f;

    const float* xr = x + (size_t)t * D_IN;
    for (int d = lane; d < D_IN; d += 32) {
        const float xv = xr[d];
        const float* g = gwm + (size_t)d * NEXP;
#pragma unroll
        for (int e = 0; e < NEXP; ++e) acc[e] += xv * g[e];
    }
#pragma unroll
    for (int e = 0; e < NEXP; ++e) {
#pragma unroll
        for (int off = 16; off > 0; off >>= 1)
            acc[e] += __shfl_down(acc[e], off, 32);
    }
    if (lane == 0) {
        int i0 = 0; float v0 = acc[0];
#pragma unroll
        for (int e = 1; e < NEXP; ++e) if (acc[e] > v0) { v0 = acc[e]; i0 = e; }
        int i1 = -1; float v1 = -3.0e38f;
#pragma unroll
        for (int e = 0; e < NEXP; ++e)
            if (e != i0 && acc[e] > v1) { v1 = acc[e]; i1 = e; }
        // softmax over the two surviving logits (mask = -1e20 elsewhere)
        const float e1 = __expf(v1 - v0);
        const float s  = 1.0f / (1.0f + e1);
        tok_e[t * 2 + 0] = i0; tok_w[t * 2 + 0] = s;
        tok_e[t * 2 + 1] = i1; tok_w[t * 2 + 1] = e1 * s;
        atomicAdd(&counts[i0], 1);
        atomicAdd(&counts[i1], 1);
    }
}

__global__ void moe_scan(const int* __restrict__ counts,
                         int* __restrict__ offsets, int* __restrict__ cursor)
{
    if (threadIdx.x == 0 && blockIdx.x == 0) {
        int s = 0;
        for (int e = 0; e < NEXP; ++e) { offsets[e] = s; s += counts[e]; cursor[e] = 0; }
        offsets[NEXP] = s;
    }
}

__global__ __launch_bounds__(256)
void moe_scatter(const int* __restrict__ tok_e, const float* __restrict__ tok_w,
                 const int* __restrict__ offsets, int* __restrict__ cursor,
                 int* __restrict__ gidx, float* __restrict__ gwt)
{
    const int t = blockIdx.x * 256 + threadIdx.x;
    if (t >= N_TOK) return;
#pragma unroll
    for (int k = 0; k < 2; ++k) {
        const int e = tok_e[t * 2 + k];
        const int p = atomicAdd(&cursor[e], 1);
        const int g = offsets[e] + p;
        gidx[g] = t;
        gwt[g]  = tok_w[t * 2 + k];
    }
}

// ---------------- phase A: act = silu(Xg*W1) * (Xg*W2) -> bf16 ----------------
// Block: 256 thr (8 waves), tile M=64 gathered rows x N=64 H-cols, K-step 32.
__global__ __launch_bounds__(256)
void moe_phaseA(const float* __restrict__ x,
                const float* __restrict__ w1,   // [D_IN, H_DIM] this expert
                const float* __restrict__ w2,
                const int* __restrict__ counts, int expert,
                const int* __restrict__ offsets,
                const int* __restrict__ gidx,
                __bf16* __restrict__ act)
{
    __shared__ __bf16 As[64 * LDT];
    __shared__ __bf16 B1[64 * LDT];
    __shared__ __bf16 B2[64 * LDT];

    const int rows = counts ? counts[expert] : N_TOK;
    const int rb = blockIdx.x * 64;
    if (rb >= rows) return;
    const int hb   = blockIdx.y * 64;
    const int base = offsets ? offsets[expert] : 0;

    const int tid  = threadIdx.x;
    const int lane = tid & 31;
    const int wave = tid >> 5;
    const int mi   = wave & 3;            // M 16-tile owned by wave
    const int ni0  = (wave >> 2) * 2;     // first of two N 16-tiles
    const int lm   = lane & 15;
    const int kbA  = (lane >> 4) * 8;
    const int kbB  = (lane >> 4) * 8;

    // A staging role: one row segment per thread
    const int ar = tid >> 2;              // 0..63
    const int ak = (tid & 3) * 8;         // 0,8,16,24
    int tok = -1;
    if (rb + ar < rows) tok = gidx ? gidx[base + rb + ar] : (rb + ar);
    const float* xrow = (tok >= 0) ? (x + (size_t)tok * D_IN) : nullptr;

    // B staging role: one k-row segment per thread
    const int wk = tid >> 3;              // 0..31
    const int wn = (tid & 7) * 8;         // 0..56

    v8f c1a = {0,0,0,0,0,0,0,0};
    v8f c1b = c1a, c2a = c1a, c2b = c1a;

    for (int k0 = 0; k0 < D_IN; k0 += 32) {
        // ---- stage A (gathered x rows, fp32 -> bf16) ----
        {
            __bf16* d = As + ar * LDT + ak;
            if (xrow) {
                const float4 f0 = *(const float4*)(xrow + k0 + ak);
                const float4 f1 = *(const float4*)(xrow + k0 + ak + 4);
                store_arow(d, f0, f1);
            } else {
                const float4 z = {0.f, 0.f, 0.f, 0.f};
                store_arow(d, z, z);
            }
        }
        // ---- stage B1/B2 transposed ([n][k], fp32 -> bf16) ----
        {
            const float* s1 = w1 + (size_t)(k0 + wk) * H_DIM + hb + wn;
            const float* s2 = w2 + (size_t)(k0 + wk) * H_DIM + hb + wn;
            const float4 a0 = *(const float4*)(s1);
            const float4 a1 = *(const float4*)(s1 + 4);
            const float4 b0 = *(const float4*)(s2);
            const float4 b1 = *(const float4*)(s2 + 4);
            store_bcol(B1 + wn * LDT + wk, a0, a1);
            store_bcol(B2 + wn * LDT + wk, b0, b1);
        }
        if (k0 + 32 < D_IN) {  // prefetch next weight K-tile -> global_prefetch
            __builtin_prefetch(w1 + (size_t)(k0 + 32 + wk) * H_DIM + hb + wn, 0, 1);
            __builtin_prefetch(w2 + (size_t)(k0 + 32 + wk) * H_DIM + hb + wn, 0, 1);
        }
        __syncthreads();

        const v16bf af  = ld_frag_a(As + (mi * 16 + lm) * LDT + kbA);
        const v16bf b1a = ld_frag_b(B1 + ((ni0    ) * 16 + lm) * LDT + kbB);
        const v16bf b1b = ld_frag_b(B1 + ((ni0 + 1) * 16 + lm) * LDT + kbB);
        const v16bf b2a = ld_frag_b(B2 + ((ni0    ) * 16 + lm) * LDT + kbB);
        const v16bf b2b = ld_frag_b(B2 + ((ni0 + 1) * 16 + lm) * LDT + kbB);

        c1a = __builtin_amdgcn_wmma_f32_16x16x32_bf16(false, af, false, b1a, (short)0, c1a, false, false);
        c1b = __builtin_amdgcn_wmma_f32_16x16x32_bf16(false, af, false, b1b, (short)0, c1b, false, false);
        c2a = __builtin_amdgcn_wmma_f32_16x16x32_bf16(false, af, false, b2a, (short)0, c2a, false, false);
        c2b = __builtin_amdgcn_wmma_f32_16x16x32_bf16(false, af, false, b2b, (short)0, c2b, false, false);
        __syncthreads();
    }

    // ---- epilogue: silu(h1)*h2 -> bf16 act[r][h] ----
    const int rr = (lane >> 4) * 8;
#pragma unroll
    for (int t2 = 0; t2 < 2; ++t2) {
        const v8f h1 = t2 ? c1b : c1a;
        const v8f h2 = t2 ? c2b : c2a;
        const int col = hb + (ni0 + t2) * 16 + lm;
#pragma unroll
        for (int i = 0; i < 8; ++i) {
            const int r = rb + mi * 16 + rr + i;
            if (r < rows) {
                const float a = h1[i];
                const float v = (a / (1.0f + __expf(-a))) * h2[i];
                act[(size_t)r * H_DIM + col] = (__bf16)v;
            }
        }
    }
}

// ---------------- phase B: out[token] (+)= weight * act * W3 ----------------
__global__ __launch_bounds__(256)
void moe_phaseB(const __bf16* __restrict__ act,
                const float* __restrict__ w3,   // [H_DIM, D_IN] this expert
                const int* __restrict__ counts, int expert,
                const int* __restrict__ offsets,
                const int* __restrict__ gidx,
                const float* __restrict__ gwt,
                float* __restrict__ out, int accumulate)
{
    __shared__ __bf16 As[64 * LDT];
    __shared__ __bf16 Bs[64 * LDT];

    const int rows = counts ? counts[expert] : N_TOK;
    const int rb = blockIdx.x * 64;
    if (rb >= rows) return;
    const int cb   = blockIdx.y * 64;
    const int base = offsets ? offsets[expert] : 0;

    const int tid  = threadIdx.x;
    const int lane = tid & 31;
    const int wave = tid >> 5;
    const int mi   = wave & 3;
    const int ni0  = (wave >> 2) * 2;
    const int lm   = lane & 15;
    const int kbA  = (lane >> 4) * 8;
    const int kbB  = (lane >> 4) * 8;

    const int ar = tid >> 2;
    const int ak = (tid & 3) * 8;
    const bool arow_ok = (rb + ar) < rows;

    const int wk = tid >> 3;
    const int wn = (tid & 7) * 8;

    // rows past the expert's count: zero their LDS segment once; the async
    // copies below only target valid rows, so these zeros persist.
    if (!arow_ok) {
        const float4 z = {0.f, 0.f, 0.f, 0.f};
        *(float4*)(As + ar * LDT + ak) = z;
    }

    v8f ca = {0,0,0,0,0,0,0,0};
    v8f cbv = ca;

    for (int k0 = 0; k0 < H_DIM; k0 += 32) {
        // ---- stage A via async global->LDS (bf16 rows, no conversion) ----
        if (arow_ok)
            async_copy_b128(act + (size_t)(rb + ar) * H_DIM + k0 + ak,
                            As + ar * LDT + ak);
        // ---- stage B transposed (w3 fp32 -> bf16), overlaps with async ----
        {
            const float* s = w3 + (size_t)(k0 + wk) * D_IN + cb + wn;
            const float4 a0 = *(const float4*)(s);
            const float4 a1 = *(const float4*)(s + 4);
            store_bcol(Bs + wn * LDT + wk, a0, a1);
        }
        if (k0 + 32 < H_DIM)
            __builtin_prefetch(w3 + (size_t)(k0 + 32 + wk) * D_IN + cb + wn, 0, 1);
        wait_async0();          // own async writes visible before barrier
        __syncthreads();

        const v16bf af = ld_frag_a(As + (mi * 16 + lm) * LDT + kbA);
        const v16bf b0 = ld_frag_b(Bs + ((ni0    ) * 16 + lm) * LDT + kbB);
        const v16bf b1 = ld_frag_b(Bs + ((ni0 + 1) * 16 + lm) * LDT + kbB);
        ca  = __builtin_amdgcn_wmma_f32_16x16x32_bf16(false, af, false, b0, (short)0, ca,  false, false);
        cbv = __builtin_amdgcn_wmma_f32_16x16x32_bf16(false, af, false, b1, (short)0, cbv, false, false);
        __syncthreads();
    }

    const int rr = (lane >> 4) * 8;
#pragma unroll
    for (int t2 = 0; t2 < 2; ++t2) {
        const v8f c = t2 ? cbv : ca;
        const int col = cb + (ni0 + t2) * 16 + lm;
#pragma unroll
        for (int i = 0; i < 8; ++i) {
            const int r = rb + mi * 16 + rr + i;
            if (r < rows) {
                const int   t = gidx ? gidx[base + r] : r;
                const float s = gwt  ? gwt[base + r]  : 1.0f;
                const size_t idx = (size_t)t * D_IN + col;
                const float v = c[i] * s;
                out[idx] = accumulate ? (out[idx] + v) : v;  // race-free per launch
            }
        }
    }
}

// ---------------- host side ----------------
extern "C" void kernel_launch(void* const* d_in, const int* in_sizes, int n_in,
                              void* d_out, int out_size, void* d_ws, size_t ws_size,
                              hipStream_t stream) {
    const float* x      = (const float*)d_in[0];
    const float* gate_w = (const float*)d_in[1];
    const float* ew1    = (const float*)d_in[2];
    const float* ew2    = (const float*)d_in[3];
    const float* ew3    = (const float*)d_in[4];
    const float* sw1    = (const float*)d_in[5];
    const float* sw2    = (const float*)d_in[6];
    const float* sw3    = (const float*)d_in[7];
    float* out = (float*)d_out;

    // workspace layout (~34.6 MB)
    char* ws = (char*)d_ws;
    int*    counts  = (int*)(ws + 0);
    int*    offsets = (int*)(ws + 256);
    int*    cursor  = (int*)(ws + 512);
    int*    tok_e   = (int*)(ws + 4096);
    float*  tok_w   = (float*)(ws + 4096 + 2 * N_TOK * 4);
    int*    gidx    = (int*)(ws + 4096 + 2 * (2 * N_TOK * 4));
    float*  gwt     = (float*)(ws + 4096 + 3 * (2 * N_TOK * 4));
    __bf16* act     = (__bf16*)(ws + (1 << 20));     // N_TOK * H_DIM bf16 = 32 MB

    hipMemsetAsync(counts, 0, NEXP * sizeof(int), stream);

    moe_gate<<<N_TOK / 8, 256, 0, stream>>>(x, gate_w, tok_e, tok_w, counts);
    moe_scan<<<1, 32, 0, stream>>>(counts, offsets, cursor);
    moe_scatter<<<N_TOK / 256, 256, 0, stream>>>(tok_e, tok_w, offsets, cursor, gidx, gwt);

    const dim3 gA(N_TOK / 64, H_DIM / 64);   // 128 x 32
    const dim3 gB(N_TOK / 64, D_IN / 64);    // 128 x 16

    // shared expert first: phase B in assign mode initializes every out element
    moe_phaseA<<<gA, 256, 0, stream>>>(x, sw1, sw2, nullptr, 0, nullptr, nullptr, act);
    moe_phaseB<<<gB, 256, 0, stream>>>(act, sw3, nullptr, 0, nullptr, nullptr, nullptr, out, 0);

    // routed experts, sequential (act buffer reuse; fixed FP summation order)
    for (int e = 0; e < NEXP; ++e) {
        const float* w1 = ew1 + (size_t)e * D_IN * H_DIM;
        const float* w2 = ew2 + (size_t)e * D_IN * H_DIM;
        const float* w3 = ew3 + (size_t)e * H_DIM * D_IN;
        moe_phaseA<<<gA, 256, 0, stream>>>(x, w1, w2, counts, e, offsets, gidx, act);
        moe_phaseB<<<gB, 256, 0, stream>>>(act, w3, counts, e, offsets, gidx, gwt, out, 1);
    }
}
